// BasicTransformerBlock_5763846111830
// MI455X (gfx1250) — matrix operand order repaired and checked by reading, not compiled, not run on hardware
//
#include <hip/hip_runtime.h>
#include <hip/hip_bf16.h>
#include <stdint.h>

// ---------------------------------------------------------------------------
// Types for CDNA5 WMMA (wave32): A/B fragments = 16 halfs, C/D = 8 floats
// ---------------------------------------------------------------------------
typedef _Float16 v16h __attribute__((ext_vector_type(16)));
typedef float    v8f  __attribute__((ext_vector_type(8)));

#define TB 256          // threads per block (8 wave32)
#define TILE_M 128
#define TILE_N 128
#define TILE_K 32
#define LDSS   40       // LDS row stride in halfs (32 data + 8 pad, 80B rows keep 16B align)

// Problem constants
#define BB     2
#define NN     4096
#define DIM    512
#define HH     8
#define DH     64
#define CTXN   77
#define CTXD   768
#define FFI    2048      // FF_INNER
#define SCALE  0.125f    // 64^-0.5

// ---------------------------------------------------------------------------
// CDNA5 async global->LDS helpers (GLOBAL_LOAD_ASYNC_TO_LDS_B128, ASYNCcnt)
// VDST VGPR holds the DS-relative LDS byte address.
// ---------------------------------------------------------------------------
__device__ __forceinline__ unsigned ldsOffsetOf(const void* p) {
    return (unsigned)(uintptr_t)(const __attribute__((address_space(3))) void*)p;
}
__device__ __forceinline__ void asyncLds16B(unsigned lds, const void* g) {
    asm volatile("global_load_async_to_lds_b128 %0, %1, off"
                 :: "v"(lds), "v"((unsigned long long)(uintptr_t)g)
                 : "memory");
}
__device__ __forceinline__ void waitAsync() {
    asm volatile("s_wait_asynccnt 0" ::: "memory");
}

// ---------------------------------------------------------------------------
// Generic NT GEMM:  C[m,n] = alpha * sum_k A[m,k] * Bt[n,k]  (+bias[n]) (+R[m,n])
// A is f16 (AF16=true) or f32 (AF16=false, converted while staging to LDS).
// Outputs: f32 C, f16 row-major O16, f16 transposed O16T (any subset).
// grid.z slices: offsets = zb*(_zb) + zh*(_zh), zb = z/Hz, zh = z%Hz.
// ---------------------------------------------------------------------------
struct GemmP {
    const _Float16* A16; const float* A32;
    const _Float16* Bt;
    const float* bias; const float* R;
    float* C; _Float16* O16; _Float16* O16T;
    int M, N, K;
    int lda, ldb, ldc, ldo, ldot, ldr;
    float alpha;
    int Hz;
    long long a_zb, a_zh, b_zb, b_zh, c_zb, c_zh;
    long long o_zb, o_zh, ot_zb, ot_zh, r_zb, r_zh;
};

union FragU { uint4 u[2]; v16h v; _Float16 h[16]; };

template<bool AF16>
__global__ __launch_bounds__(TB) void gemm_wmma_k(GemmP p) {
    __shared__ __align__(16) _Float16 As[2][TILE_M * LDSS];
    __shared__ __align__(16) _Float16 Bs[2][TILE_N * LDSS];

    const int tid  = threadIdx.x;
    const int lane = tid & 31;
    const int wave = tid >> 5;
    const int wm   = wave >> 2;     // 0..1  -> 64-row slab
    const int wn   = wave & 3;      // 0..3  -> 32-col slab
    const int m0   = blockIdx.y * TILE_M;
    const int n0   = blockIdx.x * TILE_N;
    const int zb   = (int)blockIdx.z / p.Hz;
    const int zh   = (int)blockIdx.z % p.Hz;

    const _Float16* A16 = p.A16 + zb * p.a_zb + zh * p.a_zh;
    const float*    A32 = p.A32 + zb * p.a_zb + zh * p.a_zh;
    const _Float16* Bt  = p.Bt  + zb * p.b_zb + zh * p.b_zh;

    // ---- loop-invariant per-thread staging state ----
    const int fr   = tid >> 1;            // 0..127: tile row handled by this thread
    const int koff = (tid & 1) * 16;      // 16-half chunk within the 32-wide K slab
    const int gm   = m0 + fr;
    const int gn   = n0 + fr;
    const bool aOk = gm < p.M;
    const bool bOk = gn < p.N;

    const _Float16* aRow16 = A16 + (long long)gm * p.lda + koff;
    const float*    aRow32 = A32 + (long long)gm * p.lda + koff;
    const _Float16* bRow   = Bt  + (long long)gn * p.ldb + koff;
    // Alignment is invariant: kk advances by 32 halfs (64B) / 32 floats (128B).
    const bool aFast = aOk && (AF16 ? ((((uintptr_t)aRow16) & 15u) == 0)
                                    : ((((uintptr_t)aRow32) & 15u) == 0));
    const bool bFast = bOk && ((((uintptr_t)bRow) & 15u) == 0);
    // A uses the async path only for aligned f16 sources; f32 sources need cvt.
    const bool aAsync = AF16 && aFast;

    _Float16* aSlot0 = &As[0][fr * LDSS + koff];
    _Float16* aSlot1 = &As[1][fr * LDSS + koff];
    _Float16* bSlot0 = &Bs[0][fr * LDSS + koff];
    _Float16* bSlot1 = &Bs[1][fr * LDSS + koff];
    const unsigned aL[2] = { ldsOffsetOf(aSlot0), ldsOffsetOf(aSlot1) };
    const unsigned bL[2] = { ldsOffsetOf(bSlot0), ldsOffsetOf(bSlot1) };

    // OOB rows: zero their LDS slots once (both buffers), never store again.
    {
        const uint4 z = make_uint4(0, 0, 0, 0);
        if (!aOk) {
            *(uint4*)(aSlot0) = z; *(uint4*)(aSlot0 + 8) = z;
            *(uint4*)(aSlot1) = z; *(uint4*)(aSlot1 + 8) = z;
        }
        if (!bOk) {
            *(uint4*)(bSlot0) = z; *(uint4*)(bSlot0 + 8) = z;
            *(uint4*)(bSlot1) = z; *(uint4*)(bSlot1 + 8) = z;
        }
    }

    const int kfull = p.K >> 5;                    // full 32-chunks
    const int nIter = kfull + ((p.K & 31) ? 1 : 0);

    FragU aR, bR;
    // issue async DMA for the aligned-f16 rows of K-chunk `it` into buffer `buf`
    auto fillAsync = [&](int buf, int it) {
        const bool full = (it < kfull);
        const int kk = it << 5;
        if (aAsync && full) {
            const char* g = (const char*)(aRow16 + kk);
            asyncLds16B(aL[buf],      g);
            asyncLds16B(aL[buf] + 16, g + 16);
        }
        if (bFast && full) {
            const char* g = (const char*)(bRow + kk);
            asyncLds16B(bL[buf],      g);
            asyncLds16B(bL[buf] + 16, g + 16);
        }
    };
    // register-staged loads for everything the async path can't handle
    auto loadRegsSlow = [&](int it) {
        const int kk = it << 5;
        const bool tail = (it == kfull);           // only when K%32 != 0
        if (aOk && (tail || !aAsync)) {
            if (!AF16 && aFast && !tail) {
                const float4* s = (const float4*)(aRow32 + kk);
                float4 f0 = s[0], f1 = s[1], f2 = s[2], f3 = s[3];
                aR.h[0]=(_Float16)f0.x; aR.h[1]=(_Float16)f0.y; aR.h[2]=(_Float16)f0.z; aR.h[3]=(_Float16)f0.w;
                aR.h[4]=(_Float16)f1.x; aR.h[5]=(_Float16)f1.y; aR.h[6]=(_Float16)f1.z; aR.h[7]=(_Float16)f1.w;
                aR.h[8]=(_Float16)f2.x; aR.h[9]=(_Float16)f2.y; aR.h[10]=(_Float16)f2.z; aR.h[11]=(_Float16)f2.w;
                aR.h[12]=(_Float16)f3.x; aR.h[13]=(_Float16)f3.y; aR.h[14]=(_Float16)f3.z; aR.h[15]=(_Float16)f3.w;
            } else {
                #pragma unroll
                for (int i = 0; i < 16; ++i) {
                    const int gk = kk + koff + i;
                    const bool in = tail ? (gk < p.K) : true;
                    aR.h[i] = in ? (AF16 ? aRow16[kk + i] : (_Float16)aRow32[kk + i])
                                 : (_Float16)0.f;
                }
            }
        }
        if (bOk && (tail || !bFast)) {
            #pragma unroll
            for (int i = 0; i < 16; ++i) {
                const int gk = kk + koff + i;
                const bool in = tail ? (gk < p.K) : true;
                bR.h[i] = in ? bRow[kk + i] : (_Float16)0.f;
            }
        }
    };
    auto storeRegsSlow = [&](int buf, int it) {
        const bool tail = (it == kfull);
        if (aOk && (tail || !aAsync)) {
            _Float16* d = buf ? aSlot1 : aSlot0;
            *(uint4*)(d) = aR.u[0]; *(uint4*)(d + 8) = aR.u[1];
        }
        if (bOk && (tail || !bFast)) {
            _Float16* d = buf ? bSlot1 : bSlot0;
            *(uint4*)(d) = bR.u[0]; *(uint4*)(d + 8) = bR.u[1];
        }
    };

    v8f acc[4][2] = {};
    const int frow = lane & 15;
    const int ak0  = (lane >> 4) * 8;   // A: K split in 8-chunks across lane halves
    const int bk0  = (lane >> 4) * 16;  // B: cancelled in the reads; B uses ak0 8-chunks like A

    // prologue: stage chunk 0
    fillAsync(0, 0);
    loadRegsSlow(0);
    storeRegsSlow(0, 0);
    waitAsync();
    __syncthreads();

    for (int it = 0; it < nIter; ++it) {
        const int cur = it & 1;
        const bool more = (it + 1 < nIter);
        if (more) {
            fillAsync(cur ^ 1, it + 1);        // DMA in flight during WMMA
            loadRegsSlow(it + 1);              // slow-path loads in flight too
        }

        // ---- fragments + 8 WMMAs ----
        v16h a[4], b[2];
        #pragma unroll
        for (int mi = 0; mi < 4; ++mi) {
            const _Float16* pa = &As[cur][(wm * 64 + mi * 16 + frow) * LDSS];
            FragU ua;
            ua.u[0] = *(const uint4*)(pa + ak0);          // K = ak0..ak0+7
            ua.u[1] = *(const uint4*)(pa + 16 + ak0);     // K = 16+ak0..+7
            a[mi] = ua.v;
        }
        #pragma unroll
        for (int ni = 0; ni < 2; ++ni) {
            const _Float16* pb = &Bs[cur][(wn * 32 + ni * 16 + frow) * LDSS + bk0];
            FragU ub;
            ub.u[0] = *(const uint4*)(pb - bk0 + ak0);
            ub.u[1] = *(const uint4*)(pb - bk0 + 16 + ak0);
            b[ni] = ub.v;
        }
        #pragma unroll
        for (int mi = 0; mi < 4; ++mi)
            #pragma unroll
            for (int ni = 0; ni < 2; ++ni)
                acc[mi][ni] = __builtin_amdgcn_wmma_f32_16x16x32_f16(
                    false, a[mi], false, b[ni], (short)0, acc[mi][ni], false, false);

        if (more) storeRegsSlow(cur ^ 1, it + 1);  // commit slow-path tile data
        waitAsync();                                // DMA writes visible in LDS
        __syncthreads();
    }

    // ---- epilogue ----
    const long long c_off  = (long long)zb * p.c_zb  + (long long)zh * p.c_zh;
    const long long o_off  = (long long)zb * p.o_zb  + (long long)zh * p.o_zh;
    const long long ot_off = (long long)zb * p.ot_zb + (long long)zh * p.ot_zh;
    const long long r_off  = (long long)zb * p.r_zb  + (long long)zh * p.r_zh;
    const int colHalf = lane & 15;
    const int rowHalf = 8 * (lane >> 4);
    #pragma unroll
    for (int mi = 0; mi < 4; ++mi)
        #pragma unroll
        for (int ni = 0; ni < 2; ++ni)
            #pragma unroll
            for (int r = 0; r < 8; ++r) {
                int row = m0 + wm * 64 + mi * 16 + rowHalf + r;
                int col = n0 + wn * 32 + ni * 16 + colHalf;
                if (row < p.M && col < p.N) {
                    float v = acc[mi][ni][r] * p.alpha;
                    if (p.bias) v += p.bias[col];
                    if (p.R)    v += p.R[r_off + (long long)row * p.ldr + col];
                    if (p.C)    p.C[c_off + (long long)row * p.ldc + col] = v;
                    if (p.O16)  p.O16[o_off + (long long)row * p.ldo + col] = (_Float16)v;
                    if (p.O16T) p.O16T[ot_off + (long long)col * p.ldot + row] = (_Float16)v;
                }
            }
}

// ---------------------------------------------------------------------------
// LayerNorm over last dim -> f16
// ---------------------------------------------------------------------------
__global__ __launch_bounds__(TB) void layernorm_f16(const float* __restrict__ x,
                                                    const float* __restrict__ g,
                                                    const float* __restrict__ b,
                                                    _Float16* __restrict__ out, int cols) {
    __shared__ float red[TB];
    const long long row = blockIdx.x;
    const float* p = x + row * cols;
    const int tid = threadIdx.x;
    float s = 0.f;
    for (int c = tid; c < cols; c += TB) s += p[c];
    red[tid] = s; __syncthreads();
    for (int st = TB / 2; st > 0; st >>= 1) { if (tid < st) red[tid] += red[tid + st]; __syncthreads(); }
    const float mean = red[0] / cols; __syncthreads();
    float v = 0.f;
    for (int c = tid; c < cols; c += TB) { float d = p[c] - mean; v += d * d; }
    red[tid] = v; __syncthreads();
    for (int st = TB / 2; st > 0; st >>= 1) { if (tid < st) red[tid] += red[tid + st]; __syncthreads(); }
    const float inv = rsqrtf(red[0] / cols + 1e-5f);
    _Float16* o = out + row * cols;
    for (int c = tid; c < cols; c += TB) o[c] = (_Float16)((p[c] - mean) * inv * g[c] + b[c]);
}

// ---------------------------------------------------------------------------
// In-place row softmax (one block per row)
// ---------------------------------------------------------------------------
__global__ __launch_bounds__(TB) void softmax_rows(float* __restrict__ data, int cols, long long ld) {
    __shared__ float red[TB];
    float* p = data + (long long)blockIdx.x * ld;
    const int tid = threadIdx.x;
    float m = -3.0e38f;
    for (int c = tid; c < cols; c += TB) m = fmaxf(m, p[c]);
    red[tid] = m; __syncthreads();
    for (int st = TB / 2; st > 0; st >>= 1) { if (tid < st) red[tid] = fmaxf(red[tid], red[tid + st]); __syncthreads(); }
    m = red[0]; __syncthreads();
    float s = 0.f;
    for (int c = tid; c < cols; c += TB) { float e = __expf(p[c] - m); p[c] = e; s += e; }
    red[tid] = s; __syncthreads();
    for (int st = TB / 2; st > 0; st >>= 1) { if (tid < st) red[tid] += red[tid + st]; __syncthreads(); }
    const float inv = 1.0f / red[0];
    for (int c = tid; c < cols; c += TB) p[c] *= inv;
}

// ---------------------------------------------------------------------------
// f32 [K,N] -> f16 transposed [N,K]
// ---------------------------------------------------------------------------
__global__ void tconv_f32_f16(const float* __restrict__ in, _Float16* __restrict__ out,
                              int K, int N) {
    long long idx = (long long)blockIdx.x * TB + threadIdx.x;
    if (idx < (long long)K * N) {
        int k = (int)(idx / N), n = (int)(idx % N);
        out[(long long)n * K + k] = (_Float16)in[idx];
    }
}

__global__ void conv_f32_f16(const float* __restrict__ in, _Float16* __restrict__ out, long long n) {
    long long idx = (long long)blockIdx.x * TB + threadIdx.x;
    if (idx < n) out[idx] = (_Float16)in[idx];
}

// ---------------------------------------------------------------------------
// GEGLU: h[rows,2*FFI] (f16) -> g[rows,FFI] = h[:, :FFI] * gelu_exact(h[:, FFI:])
// ---------------------------------------------------------------------------
__global__ void geglu_f16(const _Float16* __restrict__ h, _Float16* __restrict__ g, long long rows) {
    long long idx = (long long)blockIdx.x * TB + threadIdx.x;
    long long total = rows * FFI;
    if (idx < total) {
        long long m = idx / FFI; int j = (int)(idx % FFI);
        float hx   = (float)h[m * (2 * FFI) + j];
        float gate = (float)h[m * (2 * FFI) + FFI + j];
        float gl = 0.5f * gate * (1.0f + erff(gate * 0.70710678118654752f));
        g[idx] = (_Float16)(hx * gl);
    }
}

// ---------------------------------------------------------------------------
// Host
// ---------------------------------------------------------------------------
static inline void run_gemm(hipStream_t st, const GemmP& p, int Z) {
    dim3 grid((p.N + TILE_N - 1) / TILE_N, (p.M + TILE_M - 1) / TILE_M, Z);
    if (p.A16) gemm_wmma_k<true><<<grid, TB, 0, st>>>(p);
    else       gemm_wmma_k<false><<<grid, TB, 0, st>>>(p);
}

extern "C" void kernel_launch(void* const* d_in, const int* in_sizes, int n_in,
                              void* d_out, int out_size, void* d_ws, size_t ws_size,
                              hipStream_t stream) {
    (void)in_sizes; (void)n_in; (void)out_size; (void)ws_size;
    const float* x     = (const float*)d_in[0];
    const float* ctx   = (const float*)d_in[1];
    const float* ln1g  = (const float*)d_in[2];
    const float* ln1b  = (const float*)d_in[3];
    const float* ln2g  = (const float*)d_in[4];
    const float* ln2b  = (const float*)d_in[5];
    const float* ln3g  = (const float*)d_in[6];
    const float* ln3b  = (const float*)d_in[7];
    const float* wq1   = (const float*)d_in[8];
    const float* wk1   = (const float*)d_in[9];
    const float* wv1   = (const float*)d_in[10];
    const float* wo1   = (const float*)d_in[11];
    const float* bo1   = (const float*)d_in[12];
    const float* wq2   = (const float*)d_in[13];
    const float* wk2   = (const float*)d_in[14];
    const float* wv2   = (const float*)d_in[15];
    const float* wo2   = (const float*)d_in[16];
    const float* bo2   = (const float*)d_in[17];
    const float* ffw1  = (const float*)d_in[18];
    const float* ffb1  = (const float*)d_in[19];
    const float* ffw2  = (const float*)d_in[20];
    const float* ffb2  = (const float*)d_in[21];

    const long long ROWS = (long long)BB * NN;                 // 8192
    float* out_x     = (float*)d_out;                          // [8192,512]
    float* self_map  = out_x + ROWS * DIM;                     // [16,4096,4096]
    float* cross_map = self_map + (long long)BB * HH * NN * NN;// [16,4096,77]

    // ---- workspace carve ----
    char* w = (char*)d_ws;
    auto alloc = [&](size_t bytes) {
        void* p = (void*)w; w += (bytes + 255) & ~(size_t)255; return p;
    };
    _Float16* wq1t  = (_Float16*)alloc((size_t)DIM * DIM * 2);
    _Float16* wk1t  = (_Float16*)alloc((size_t)DIM * DIM * 2);
    _Float16* wv1t  = (_Float16*)alloc((size_t)DIM * DIM * 2);
    _Float16* wo1t  = (_Float16*)alloc((size_t)DIM * DIM * 2);
    _Float16* wq2t  = (_Float16*)alloc((size_t)DIM * DIM * 2);
    _Float16* wk2t  = (_Float16*)alloc((size_t)DIM * CTXD * 2);
    _Float16* wv2t  = (_Float16*)alloc((size_t)DIM * CTXD * 2);
    _Float16* wo2t  = (_Float16*)alloc((size_t)DIM * DIM * 2);
    _Float16* wff1t = (_Float16*)alloc((size_t)(2 * FFI) * DIM * 2);
    _Float16* wff2t = (_Float16*)alloc((size_t)DIM * FFI * 2);
    _Float16* ctx16 = (_Float16*)alloc((size_t)BB * CTXN * CTXD * 2);
    _Float16* lnb   = (_Float16*)alloc((size_t)ROWS * DIM * 2);
    _Float16* q1    = (_Float16*)alloc((size_t)ROWS * DIM * 2);   // reused as q2
    _Float16* k1    = (_Float16*)alloc((size_t)ROWS * DIM * 2);
    _Float16* vt    = (_Float16*)alloc((size_t)BB * DIM * NN * 2);  // V^T per batch [512,4096]
    _Float16* ao    = (_Float16*)alloc((size_t)ROWS * DIM * 2);   // attention outputs
    _Float16* k2    = (_Float16*)alloc((size_t)BB * CTXN * DIM * 2);
    _Float16* vt2   = (_Float16*)alloc((size_t)BB * DIM * CTXN * 2); // V2^T per batch [512,77]
    _Float16* h16   = (_Float16*)alloc((size_t)ROWS * 2 * FFI * 2);
    _Float16* g16   = (_Float16*)alloc((size_t)ROWS * FFI * 2);
    float*    x1    = (float*)alloc((size_t)ROWS * DIM * 4);
    float*    x2    = (float*)alloc((size_t)ROWS * DIM * 4);

    auto gblk = [](long long n) { return (unsigned)((n + TB - 1) / TB); };

    // ---- weight transpose+convert ----
    tconv_f32_f16<<<gblk((long long)DIM * DIM), TB, 0, stream>>>(wq1, wq1t, DIM, DIM);
    tconv_f32_f16<<<gblk((long long)DIM * DIM), TB, 0, stream>>>(wk1, wk1t, DIM, DIM);
    tconv_f32_f16<<<gblk((long long)DIM * DIM), TB, 0, stream>>>(wv1, wv1t, DIM, DIM);
    tconv_f32_f16<<<gblk((long long)DIM * DIM), TB, 0, stream>>>(wo1, wo1t, DIM, DIM);
    tconv_f32_f16<<<gblk((long long)DIM * DIM), TB, 0, stream>>>(wq2, wq2t, DIM, DIM);
    tconv_f32_f16<<<gblk((long long)CTXD * DIM), TB, 0, stream>>>(wk2, wk2t, CTXD, DIM);
    tconv_f32_f16<<<gblk((long long)CTXD * DIM), TB, 0, stream>>>(wv2, wv2t, CTXD, DIM);
    tconv_f32_f16<<<gblk((long long)DIM * DIM), TB, 0, stream>>>(wo2, wo2t, DIM, DIM);
    tconv_f32_f16<<<gblk((long long)DIM * 2 * FFI), TB, 0, stream>>>(ffw1, wff1t, DIM, 2 * FFI);
    tconv_f32_f16<<<gblk((long long)FFI * DIM), TB, 0, stream>>>(ffw2, wff2t, FFI, DIM);
    conv_f32_f16<<<gblk((long long)BB * CTXN * CTXD), TB, 0, stream>>>(ctx, ctx16, (long long)BB * CTXN * CTXD);

    GemmP p;
    auto base = [&]() { GemmP q{}; q.alpha = 1.0f; q.Hz = 1; return q; };

    // ============== self-attention ==============
    layernorm_f16<<<(unsigned)ROWS, TB, 0, stream>>>(x, ln1g, ln1b, lnb, DIM);

    p = base(); p.A16 = lnb; p.Bt = wq1t; p.O16 = q1;
    p.M = (int)ROWS; p.N = DIM; p.K = DIM; p.lda = DIM; p.ldb = DIM; p.ldo = DIM;
    run_gemm(stream, p, 1);

    p.Bt = wk1t; p.O16 = k1; run_gemm(stream, p, 1);

    // V projection, stored transposed per batch: vt[b][d][j] = V[b][j][d]
    p = base(); p.A16 = lnb; p.Bt = wv1t; p.O16T = vt;
    p.M = NN; p.N = DIM; p.K = DIM; p.lda = DIM; p.ldb = DIM; p.ldot = NN;
    p.a_zb = (long long)NN * DIM; p.ot_zb = (long long)DIM * NN;
    run_gemm(stream, p, BB);

    // S = scale * Q Kt  -> self_map  (z = b*H + h)
    p = base(); p.Hz = HH; p.alpha = SCALE;
    p.A16 = q1; p.lda = DIM; p.a_zb = (long long)NN * DIM; p.a_zh = DH;
    p.Bt  = k1; p.ldb = DIM; p.b_zb = (long long)NN * DIM; p.b_zh = DH;
    p.C = self_map; p.ldc = NN;
    p.c_zb = (long long)HH * NN * NN; p.c_zh = (long long)NN * NN;
    p.M = NN; p.N = NN; p.K = DH;
    run_gemm(stream, p, BB * HH);

    softmax_rows<<<(unsigned)(BB * HH * NN), TB, 0, stream>>>(self_map, NN, NN);

    // out_h = P @ V   (A is f32 from d_out, converted on the fly)
    p = base(); p.Hz = HH;
    p.A32 = self_map; p.lda = NN;
    p.a_zb = (long long)HH * NN * NN; p.a_zh = (long long)NN * NN;
    p.Bt = vt; p.ldb = NN; p.b_zb = (long long)DIM * NN; p.b_zh = (long long)DH * NN;
    p.O16 = ao; p.ldo = DIM; p.o_zb = (long long)NN * DIM; p.o_zh = DH;
    p.M = NN; p.N = DH; p.K = NN;
    run_gemm(stream, p, BB * HH);

    // x1 = x + ao @ Wo1 + bo1
    p = base(); p.A16 = ao; p.Bt = wo1t; p.bias = bo1; p.R = x; p.ldr = DIM;
    p.C = x1; p.ldc = DIM; p.M = (int)ROWS; p.N = DIM; p.K = DIM; p.lda = DIM; p.ldb = DIM;
    run_gemm(stream, p, 1);

    // ============== cross-attention ==============
    layernorm_f16<<<(unsigned)ROWS, TB, 0, stream>>>(x1, ln2g, ln2b, lnb, DIM);

    p = base(); p.A16 = lnb; p.Bt = wq2t; p.O16 = q1;   // reuse q1 buffer as q2
    p.M = (int)ROWS; p.N = DIM; p.K = DIM; p.lda = DIM; p.ldb = DIM; p.ldo = DIM;
    run_gemm(stream, p, 1);

    p = base(); p.A16 = ctx16; p.Bt = wk2t; p.O16 = k2;
    p.M = BB * CTXN; p.N = DIM; p.K = CTXD; p.lda = CTXD; p.ldb = CTXD; p.ldo = DIM;
    run_gemm(stream, p, 1);

    p = base(); p.A16 = ctx16; p.Bt = wv2t; p.O16T = vt2;
    p.M = CTXN; p.N = DIM; p.K = CTXD; p.lda = CTXD; p.ldb = CTXD; p.ldot = CTXN;
    p.a_zb = (long long)CTXN * CTXD; p.ot_zb = (long long)DIM * CTXN;
    run_gemm(stream, p, BB);

    // S2 = scale * Q2 K2t -> cross_map
    p = base(); p.Hz = HH; p.alpha = SCALE;
    p.A16 = q1; p.lda = DIM; p.a_zb = (long long)NN * DIM; p.a_zh = DH;
    p.Bt  = k2; p.ldb = DIM; p.b_zb = (long long)CTXN * DIM; p.b_zh = DH;
    p.C = cross_map; p.ldc = CTXN;
    p.c_zb = (long long)HH * NN * CTXN; p.c_zh = (long long)NN * CTXN;
    p.M = NN; p.N = CTXN; p.K = DH;
    run_gemm(stream, p, BB * HH);

    softmax_rows<<<(unsigned)(BB * HH * NN), TB, 0, stream>>>(cross_map, CTXN, CTXN);

    // out_h = P2 @ V2
    p = base(); p.Hz = HH;
    p.A32 = cross_map; p.lda = CTXN;
    p.a_zb = (long long)HH * NN * CTXN; p.a_zh = (long long)NN * CTXN;
    p.Bt = vt2; p.ldb = CTXN; p.b_zb = (long long)DIM * CTXN; p.b_zh = (long long)DH * CTXN;
    p.O16 = ao; p.ldo = DIM; p.o_zb = (long long)NN * DIM; p.o_zh = DH;
    p.M = NN; p.N = DH; p.K = CTXN;
    run_gemm(stream, p, BB * HH);

    // x2 = x1 + ao @ Wo2 + bo2
    p = base(); p.A16 = ao; p.Bt = wo2t; p.bias = bo2; p.R = x1; p.ldr = DIM;
    p.C = x2; p.ldc = DIM; p.M = (int)ROWS; p.N = DIM; p.K = DIM; p.lda = DIM; p.ldb = DIM;
    run_gemm(stream, p, 1);

    // ============== GEGLU feed-forward ==============
    layernorm_f16<<<(unsigned)ROWS, TB, 0, stream>>>(x2, ln3g, ln3b, lnb, DIM);

    p = base(); p.A16 = lnb; p.Bt = wff1t; p.bias = ffb1; p.O16 = h16;
    p.M = (int)ROWS; p.N = 2 * FFI; p.K = DIM; p.lda = DIM; p.ldb = DIM; p.ldo = 2 * FFI;
    run_gemm(stream, p, 1);

    geglu_f16<<<gblk(ROWS * FFI), TB, 0, stream>>>(h16, g16, ROWS);

    p = base(); p.A16 = g16; p.Bt = wff2t; p.bias = ffb2; p.R = x2; p.ldr = DIM;
    p.C = out_x; p.ldc = DIM; p.M = (int)ROWS; p.N = DIM; p.K = FFI; p.lda = FFI; p.ldb = FFI;
    run_gemm(stream, p, 1);
}
